// SingleHeadAttention_16664473108480
// MI455X (gfx1250) — matrix operand order repaired and checked by reading, not compiled, not run on hardware
//
#include <hip/hip_runtime.h>
#include <cstdint>

typedef __bf16 bf16_t;
typedef __attribute__((ext_vector_type(16))) __bf16 v16bf;
typedef __attribute__((ext_vector_type(8)))  __bf16 v8bf;
typedef __attribute__((ext_vector_type(8)))  float   v8f;
typedef __attribute__((ext_vector_type(4)))  float   v4f;

#define TSEQ 4096
#define DIM  1024
#define SCALE 0.03125f  /* 1/sqrt(1024) */

static __device__ __forceinline__ v8f wmma_bf16(v16bf a, v16bf b, v8f c) {
  return __builtin_amdgcn_wmma_f32_16x16x32_bf16(false, a, false, b, (short)0, c,
                                                 false, false);
}

static __device__ __forceinline__ void splitf(float x, bf16_t& hi, bf16_t& lo) {
  hi = (bf16_t)x;
  lo = (bf16_t)(x - (float)hi);
}

static __device__ __forceinline__ v16bf cat8(v8bf a, v8bf b) {
  v16bf r;
#pragma unroll
  for (int i = 0; i < 8; ++i) { r[i] = a[i]; r[i + 8] = b[i]; }
  return r;
}

// Async DMA: global -> LDS, 16 bytes, tracked by ASYNCcnt.
static __device__ __forceinline__ void async_b128(uint32_t lds_addr, const void* gaddr) {
  asm volatile("global_load_async_to_lds_b128 %0, %1, off"
               :: "v"(lds_addr), "v"((uint64_t)(uintptr_t)gaddr)
               : "memory");
}
static __device__ __forceinline__ void wait_async0() {
#if __has_builtin(__builtin_amdgcn_s_wait_asynccnt)
  __builtin_amdgcn_s_wait_asynccnt(0);
#else
  asm volatile("s_wait_asynccnt 0" ::: "memory");
#endif
}
static __device__ __forceinline__ uint32_t lds_addr_of(const void* p) {
  return (uint32_t)(uintptr_t)p;
}

// ---------------------------------------------------------------------------
// Kernel 1: [16384,1024] @ [1024,1024] -> bf16 hi/lo planes, with register
// prefetch of the next K-tile overlapped against the WMMA block.
// transposed=1 writes Y^T per batch ([b][col][row]) to give K^T layout.
// ---------------------------------------------------------------------------
__global__ __launch_bounds__(256) void qkv_gemm(const float* __restrict__ X,
                                                const float* __restrict__ W,
                                                bf16_t* __restrict__ Yhi,
                                                bf16_t* __restrict__ Ylo,
                                                int transposed) {
  __shared__ __align__(16) bf16_t sAhi[128 * 40];
  __shared__ __align__(16) bf16_t sAlo[128 * 40];
  __shared__ __align__(16) bf16_t sBhi[32 * 136];
  __shared__ __align__(16) bf16_t sBlo[32 * 136];

  const int tid = threadIdx.x;
  const int wv = tid >> 5, lane = tid & 31;
  const int n0 = blockIdx.x * 128;
  const int m0 = blockIdx.y * 128;
  const int wr = (wv >> 1) * 32;
  const int wc = (wv & 1) * 64;
  const int hb = (lane >= 16) ? 8 : 0;
  const int kb8 = (lane < 16) ? 0 : 8;

  v8f acc[2][4];
#pragma unroll
  for (int i = 0; i < 2; ++i)
#pragma unroll
    for (int j = 0; j < 4; ++j)
#pragma unroll
      for (int e = 0; e < 8; ++e) acc[i][j][e] = 0.0f;

  const int xr = tid >> 3, xc = (tid & 7) * 4;
  const int wk = tid >> 5, wn = (tid & 31) * 4;

  v4f xa[4], wb[4];
#pragma unroll
  for (int p = 0; p < 4; ++p) {          // prefetch tile 0
    xa[p] = *(const v4f*)(X + (size_t)(m0 + xr + p * 32) * DIM + xc);
    wb[p] = *(const v4f*)(W + (size_t)(wk + p * 8) * DIM + n0 + wn);
  }

  for (int ks = 0; ks < 32; ++ks) {
    __syncthreads();
#pragma unroll
    for (int p = 0; p < 4; ++p) {        // spill prefetched tile to LDS hi/lo
      int r = xr + p * 32;
#pragma unroll
      for (int j = 0; j < 4; ++j)
        splitf(xa[p][j], sAhi[r * 40 + xc + j], sAlo[r * 40 + xc + j]);
      int rw = wk + p * 8;
#pragma unroll
      for (int j = 0; j < 4; ++j)
        splitf(wb[p][j], sBhi[rw * 136 + wn + j], sBlo[rw * 136 + wn + j]);
    }
    __syncthreads();
    if (ks < 31) {                       // prefetch next tile during compute
      const int k0 = (ks + 1) * 32;
#pragma unroll
      for (int p = 0; p < 4; ++p) {
        xa[p] = *(const v4f*)(X + (size_t)(m0 + xr + p * 32) * DIM + k0 + xc);
        wb[p] = *(const v4f*)(W + (size_t)(k0 + wk + p * 8) * DIM + n0 + wn);
      }
    }

    v16bf ahi[2], alo[2], bhi[4], blo[4];
    const int arow = wr + (lane & 15);
#pragma unroll
    for (int ti = 0; ti < 2; ++ti) {
      const bf16_t* ph = &sAhi[(arow + ti * 16) * 40 + kb8];
      const bf16_t* pl = &sAlo[(arow + ti * 16) * 40 + kb8];
      ahi[ti] = cat8(*(const v8bf*)ph, *(const v8bf*)(ph + 16));
      alo[ti] = cat8(*(const v8bf*)pl, *(const v8bf*)(pl + 16));
    }
#pragma unroll
    for (int tj = 0; tj < 4; ++tj) {
#pragma unroll
      for (int s = 0; s < 16; ++s) {
        const int kk = kb8 + (s & 7) + (s >> 3) * 16;
        bhi[tj][s] = sBhi[kk * 136 + wc + tj * 16 + (lane & 15)];
        blo[tj][s] = sBlo[kk * 136 + wc + tj * 16 + (lane & 15)];
      }
    }
#pragma unroll
    for (int ti = 0; ti < 2; ++ti)
#pragma unroll
      for (int tj = 0; tj < 4; ++tj) {
        acc[ti][tj] = wmma_bf16(ahi[ti], bhi[tj], acc[ti][tj]);
        acc[ti][tj] = wmma_bf16(ahi[ti], blo[tj], acc[ti][tj]);
        acc[ti][tj] = wmma_bf16(alo[ti], bhi[tj], acc[ti][tj]);
      }
  }
#pragma unroll
  for (int ti = 0; ti < 2; ++ti)
#pragma unroll
    for (int tj = 0; tj < 4; ++tj)
#pragma unroll
      for (int j = 0; j < 8; ++j) {
        int row = m0 + wr + ti * 16 + j + hb;
        int col = n0 + wc + tj * 16 + (lane & 15);
        bf16_t h, o;
        splitf(acc[ti][tj][j], h, o);
        size_t idx = transposed
            ? ((size_t)(row >> 12) * DIM + col) * TSEQ + (row & (TSEQ - 1))
            : (size_t)row * DIM + col;
        Yhi[idx] = h;
        Ylo[idx] = o;
      }
}

// ---------------------------------------------------------------------------
// Kernel 2: per-row softmax stats m, l. 128 queries / block, 8 waves x 16 rows.
// K^T staged via double-buffered async LDS copies (DMA overlapped w/ WMMA).
// ---------------------------------------------------------------------------
__global__ __launch_bounds__(256) void attn_stats(const bf16_t* __restrict__ Qhi,
                                                  const bf16_t* __restrict__ Qlo,
                                                  const bf16_t* __restrict__ KThi,
                                                  const bf16_t* __restrict__ KTlo,
                                                  float* __restrict__ Ms,
                                                  float* __restrict__ Ls) {
  // 2 ping-pong buffers x (hi,lo planes) x [32 a][136 keys-padded]
  __shared__ __align__(16) bf16_t sK[2 * 2 * 4352];
  const int tid = threadIdx.x, wv = tid >> 5, lane = tid & 31;
  const int b = blockIdx.y;
  const int Q0 = blockIdx.x * 128;
  const int hb = (lane >= 16) ? 8 : 0;
  const int kb8 = (lane < 16) ? 0 : 8;
  const size_t qbase = ((size_t)b * TSEQ + Q0 + wv * 16 + (lane & 15)) * DIM;

  float m[8], l[8];
#pragma unroll
  for (int j = 0; j < 8; ++j) { m[j] = -3.0e38f; l[j] = 0.0f; }

  const int sj = tid >> 3, scl = (tid & 7) * 16;  // staging map: 32 rows x 8 thr

  auto stage = [&](int kb, int ac, int bb) {
    const size_t gro = ((size_t)b * DIM + ac * 32 + sj) * TSEQ + kb * 128 + scl;
    uint32_t dh = lds_addr_of(&sK[bb * 8704 + sj * 136 + scl]);
    uint32_t dl = dh + 4352 * 2;
    async_b128(dh,      KThi + gro);
    async_b128(dh + 16, KThi + gro + 8);
    async_b128(dl,      KTlo + gro);
    async_b128(dl + 16, KTlo + gro + 8);
  };

  for (int kb = 0; kb <= (int)blockIdx.x; ++kb) {
    v8f acc[8];
#pragma unroll
    for (int t = 0; t < 8; ++t)
#pragma unroll
      for (int e = 0; e < 8; ++e) acc[t][e] = 0.0f;

    stage(kb, 0, 0);
    for (int ac = 0; ac < 32; ++ac) {
      wait_async0();
      __syncthreads();
      if (ac < 31) stage(kb, ac + 1, (ac + 1) & 1);
      const int bb = ac & 1;

      const int a0 = ac * 32 + kb8;
      const bf16_t* qh = Qhi + qbase + a0;
      const bf16_t* ql = Qlo + qbase + a0;
      v16bf qfh = cat8(*(const v8bf*)qh, *(const v8bf*)(qh + 16));
      v16bf qfl = cat8(*(const v8bf*)ql, *(const v8bf*)(ql + 16));
#pragma unroll
      for (int tj = 0; tj < 8; ++tj) {
        v16bf khi, klo;
#pragma unroll
        for (int s = 0; s < 16; ++s) {
          const int kk = kb8 + (s & 7) + (s >> 3) * 16;
          const bf16_t* ph = &sK[bb * 8704 + kk * 136 + tj * 16 + (lane & 15)];
          khi[s] = ph[0];
          klo[s] = ph[4352];
        }
        acc[tj] = wmma_bf16(qfh, khi, acc[tj]);
        acc[tj] = wmma_bf16(qfh, klo, acc[tj]);
        acc[tj] = wmma_bf16(qfl, khi, acc[tj]);
      }
    }

    const int qg0 = Q0 + wv * 16 + hb;   // online softmax update
#pragma unroll
    for (int j = 0; j < 8; ++j) {
      int qg = qg0 + j;
      float bm = -3.0e38f;
#pragma unroll
      for (int tj = 0; tj < 8; ++tj) {
        int kg = kb * 128 + tj * 16 + (lane & 15);
        float s = (kg <= qg) ? acc[tj][j] * SCALE : -3.0e38f;
        acc[tj][j] = s;
        bm = fmaxf(bm, s);
      }
#pragma unroll
      for (int off = 8; off >= 1; off >>= 1)
        bm = fmaxf(bm, __shfl_xor(bm, off, 32));
      float mn = fmaxf(m[j], bm);
      float sum = 0.0f;
#pragma unroll
      for (int tj = 0; tj < 8; ++tj) sum += __expf(acc[tj][j] - mn);
#pragma unroll
      for (int off = 8; off >= 1; off >>= 1) sum += __shfl_xor(sum, off, 32);
      l[j] = l[j] * __expf(m[j] - mn) + sum;
      m[j] = mn;
    }
  }
#pragma unroll
  for (int j = 0; j < 8; ++j) {
    if ((lane & 15) == j) {
      int qg = Q0 + wv * 16 + j + hb;
      Ms[(size_t)b * TSEQ + qg] = m[j];
      Ls[(size_t)b * TSEQ + qg] = l[j];
    }
  }
}

// ---------------------------------------------------------------------------
// Kernel 3: O = softmax(QK^T) V with precomputed m,l. 64 queries / block,
// 4 waves x 16 rows; O in LDS (64x1028 f32). Double-buffered async staging;
// K^T staging buffers alias the P region (disjoint phases, barrier handoff).
// LDS: 263168 (O) + 34816 (P / K^T bufs) + 24576 (V bufs) = 322560 B.
// ---------------------------------------------------------------------------
__global__ __launch_bounds__(128) void attn_out(const bf16_t* __restrict__ Qhi,
                                                const bf16_t* __restrict__ Qlo,
                                                const bf16_t* __restrict__ KThi,
                                                const bf16_t* __restrict__ KTlo,
                                                const bf16_t* __restrict__ Vhi,
                                                const bf16_t* __restrict__ Vlo,
                                                const float* __restrict__ Ms,
                                                const float* __restrict__ Ls,
                                                float* __restrict__ Out) {
  extern __shared__ __align__(16) char smem[];
  float*  sO = (float*)smem;                      // 64 x 1028 f32
  // P region: wave w -> [w*4352 .. ], planes at +0 / +2176 (16x136 each).
  // Phase 1 aliases this as K^T ping-pong: buf bb at bb*8704, planes +0/+4352.
  bf16_t* sP = (bf16_t*)(smem + 64 * 1028 * 4);   // 17408 elems
  // V ping-pong: buf bb at bb*6144, planes +0/+3072, [128 key][24 pad] x 16 a.
  bf16_t* sV = sP + 17408;                        // 12288 elems

  const int tid = threadIdx.x, wv = tid >> 5, lane = tid & 31;
  const int b = blockIdx.y;
  const int Q0 = blockIdx.x * 64;
  const int wrow = wv * 16;
  const int hb = (lane >= 16) ? 8 : 0;
  const int kb8 = (lane < 16) ? 0 : 8;

  for (int i = tid; i < 64 * 1028; i += 128) sO[i] = 0.0f;

  float m[8];
#pragma unroll
  for (int j = 0; j < 8; ++j)
    m[j] = Ms[(size_t)b * TSEQ + Q0 + wrow + j + hb];

  const size_t qbase = ((size_t)b * TSEQ + Q0 + wrow + (lane & 15)) * DIM;
  const int kbmax = blockIdx.x >> 1;
  __syncthreads();

  auto stage_k = [&](int kb, int ac, int bb) {   // 128 thr = 32 rows x 4
    const int sj = tid >> 2, scl = (tid & 3) * 32;
    const size_t gro = ((size_t)b * DIM + ac * 32 + sj) * TSEQ + kb * 128 + scl;
    uint32_t dh = lds_addr_of(&sP[bb * 8704 + sj * 136 + scl]);
    uint32_t dl = dh + 4352 * 2;
#pragma unroll
    for (int c = 0; c < 4; ++c) {
      async_b128(dh + c * 16, KThi + gro + c * 8);
      async_b128(dl + c * 16, KTlo + gro + c * 8);
    }
  };
  auto stage_v = [&](int kb, int a2, int bb) {   // 1 key per thread, 16 a-cols
    const size_t gro = ((size_t)b * TSEQ + kb * 128 + tid) * DIM + a2 * 16;
    uint32_t dh = lds_addr_of(&sV[bb * 6144 + tid * 24]);
    uint32_t dl = dh + 3072 * 2;
    async_b128(dh,      Vhi + gro);
    async_b128(dh + 16, Vhi + gro + 8);
    async_b128(dl,      Vlo + gro);
    async_b128(dl + 16, Vlo + gro + 8);
  };

  for (int kb = 0; kb <= kbmax; ++kb) {
    // ---- phase 1: recompute score tile (16 x 128 per wave)
    v8f acc[8];
#pragma unroll
    for (int t = 0; t < 8; ++t)
#pragma unroll
      for (int e = 0; e < 8; ++e) acc[t][e] = 0.0f;

    stage_k(kb, 0, 0);
    for (int ac = 0; ac < 32; ++ac) {
      wait_async0();
      __syncthreads();
      if (ac < 31) stage_k(kb, ac + 1, (ac + 1) & 1);
      const int bb = ac & 1;

      const int a0 = ac * 32 + kb8;
      const bf16_t* qh = Qhi + qbase + a0;
      const bf16_t* ql = Qlo + qbase + a0;
      v16bf qfh = cat8(*(const v8bf*)qh, *(const v8bf*)(qh + 16));
      v16bf qfl = cat8(*(const v8bf*)ql, *(const v8bf*)(ql + 16));
#pragma unroll
      for (int tj = 0; tj < 8; ++tj) {
        v16bf khi, klo;
#pragma unroll
        for (int s = 0; s < 16; ++s) {
          const int kk = kb8 + (s & 7) + (s >> 3) * 16;
          const bf16_t* ph = &sP[bb * 8704 + kk * 136 + tj * 16 + (lane & 15)];
          khi[s] = ph[0];
          klo[s] = ph[4352];
        }
        acc[tj] = wmma_bf16(qfh, khi, acc[tj]);
        acc[tj] = wmma_bf16(qfh, klo, acc[tj]);
        acc[tj] = wmma_bf16(qfl, khi, acc[tj]);
      }
    }
    __syncthreads();   // hand K^T staging region back to P before writing it

    // ---- phase 2: P = exp(s - m), masked, into per-wave LDS (A layout)
    const int qg0 = Q0 + wrow + hb;
#pragma unroll
    for (int tj = 0; tj < 8; ++tj)
#pragma unroll
      for (int j = 0; j < 8; ++j) {
        int kg = kb * 128 + tj * 16 + (lane & 15);
        float s = (kg <= qg0 + j) ? acc[tj][j] * SCALE : -3.0e38f;
        float pv = __expf(s - m[j]);
        int idx = wv * 4352 + (j + hb) * 136 + tj * 16 + (lane & 15);
        splitf(pv, sP[idx], sP[idx + 2176]);
      }

    // ---- phase 3: O += P @ V, 16-column chunks, C operand from LDS
    stage_v(kb, 0, 0);
    for (int a2 = 0; a2 < 64; ++a2) {
      wait_async0();
      __syncthreads();
      if (a2 < 63) stage_v(kb, a2 + 1, (a2 + 1) & 1);
      const int bb = a2 & 1;

      v8f o;
#pragma unroll
      for (int j = 0; j < 8; ++j)
        o[j] = sO[(wrow + j + hb) * 1028 + a2 * 16 + (lane & 15)];
#pragma unroll
      for (int ksq = 0; ksq < 4; ++ksq) {
        const bf16_t* pp = &sP[wv * 4352 + (lane & 15) * 136 + ksq * 32 + kb8];
        v16bf phi = cat8(*(const v8bf*)pp, *(const v8bf*)(pp + 16));
        v16bf plo = cat8(*(const v8bf*)(pp + 2176), *(const v8bf*)(pp + 2176 + 16));
        v16bf vhi, vlo;
#pragma unroll
        for (int s = 0; s < 16; ++s) {
          const int kk = kb8 + (s & 7) + (s >> 3) * 16;
          const bf16_t* vh = &sV[bb * 6144 + (ksq * 32 + kk) * 24 + (lane & 15)];
          vhi[s] = vh[0];
          vlo[s] = vh[3072];
        }
        o = wmma_bf16(phi, vhi, o);
        o = wmma_bf16(phi, vlo, o);
        o = wmma_bf16(plo, vhi, o);
      }
#pragma unroll
      for (int j = 0; j < 8; ++j)
        sO[(wrow + j + hb) * 1028 + a2 * 16 + (lane & 15)] = o[j];
    }
    __syncthreads();   // end of kb: P region reused as K^T buffers next round
  }

  for (int i = tid; i < 64 * DIM; i += 128) {
    int row = i >> 10, col = i & (DIM - 1);
    float lv = Ls[(size_t)b * TSEQ + Q0 + row];
    Out[((size_t)b * TSEQ + Q0 + row) * DIM + col] = sO[row * 1028 + col] / lv;
  }
}

// ---------------------------------------------------------------------------
extern "C" void kernel_launch(void* const* d_in, const int* in_sizes, int n_in,
                              void* d_out, int out_size, void* d_ws, size_t ws_size,
                              hipStream_t stream) {
  (void)in_sizes; (void)n_in; (void)out_size; (void)ws_size;
  const float* X  = (const float*)d_in[0];
  const float* Wk = (const float*)d_in[1];
  const float* Wq = (const float*)d_in[2];
  const float* Wv = (const float*)d_in[3];

  bf16_t* base = (bf16_t*)d_ws;
  const size_t PS = (size_t)16384 * DIM;   // plane size (elements)
  bf16_t* Qhi  = base;
  bf16_t* Qlo  = base + PS;
  bf16_t* KThi = base + 2 * PS;            // K stored transposed: [b][a][t]
  bf16_t* KTlo = base + 3 * PS;
  bf16_t* Vhi  = base + 4 * PS;
  bf16_t* Vlo  = base + 5 * PS;
  float* Ms = (float*)(base + 6 * PS);
  float* Ls = Ms + 16384;
  float* Out = (float*)d_out;

  dim3 g1(DIM / 128, 16384 / 128);         // (8, 128)
  qkv_gemm<<<g1, 256, 0, stream>>>(X, Wq, Qhi, Qlo, 0);
  qkv_gemm<<<g1, 256, 0, stream>>>(X, Wk, KThi, KTlo, 1);
  qkv_gemm<<<g1, 256, 0, stream>>>(X, Wv, Vhi, Vlo, 0);

  attn_stats<<<dim3(TSEQ / 128, 4), 256, 0, stream>>>(Qhi, Qlo, KThi, KTlo, Ms, Ls);

  size_t shmem = (size_t)64 * 1028 * 4     // O accumulator
               + (size_t)17408 * 2         // P region / K^T ping-pong (aliased)
               + (size_t)12288 * 2;        // V ping-pong
  attn_out<<<dim3(TSEQ / 64, 4), 128, shmem, stream>>>(Qhi, Qlo, KThi, KTlo,
                                                       Vhi, Vlo, Ms, Ls, Out);
}
